// Attention_32873679684330
// MI455X (gfx1250) — matrix operand order repaired and checked by reading, not compiled, not run on hardware
//
#include <hip/hip_runtime.h>
#include <math.h>

typedef __attribute__((ext_vector_type(16))) __bf16 bf16x16;
typedef __attribute__((ext_vector_type(8)))  __bf16 bf16x8;
typedef __attribute__((ext_vector_type(8)))  float  f32x8;
typedef __attribute__((ext_vector_type(4)))  int    i32x4;

// Problem constants
#define BB   8
#define PP   1024
#define EE   1024
#define HH   16
#define DHD  64
#define MLPD 4096
#define ROWS (BB*PP)          // 8192
#define ATTN_SCALE 0.125f     // DH^-0.5

// ---- gfx1250 async global->LDS path (guarded; falls back to reg staging) ----
#if defined(__has_builtin)
#if __has_builtin(__builtin_amdgcn_global_load_async_to_lds_b128) && \
    __has_builtin(__builtin_amdgcn_s_wait_asynccnt)
#define USE_ASYNC_LDS 1
#endif
#endif

#ifdef USE_ASYNC_LDS
static __device__ inline void cp_async16(void* lds, const void* glb) {
    __builtin_amdgcn_global_load_async_to_lds_b128(
        (__attribute__((address_space(1))) i32x4*)glb,
        (__attribute__((address_space(3))) i32x4*)lds, 0, 0);
}
static __device__ inline void async_wait0() { __builtin_amdgcn_s_wait_asynccnt(0); }
#endif

static __device__ inline f32x8 wmma_bf16(bf16x16 a, bf16x16 b, f32x8 c) {
    return __builtin_amdgcn_wmma_f32_16x16x32_bf16(false, a, false, b, (short)0, c, false, false);
}

static __device__ inline bf16x16 ld16_pair(const __bf16* plo, const __bf16* phi) {
    bf16x8 lo = *(const bf16x8*)plo;
    bf16x8 hi = *(const bf16x8*)phi;
    return __builtin_shufflevector(lo, hi, 0,1,2,3,4,5,6,7,8,9,10,11,12,13,14,15);
}

// ------------------------------------------------ fp32 [K][N] -> bf16 [N][K]
// tiled 32x32 transpose, coalesced reads and writes
__global__ __launch_bounds__(256) void cvt_t_bf16(const float* __restrict__ in,
                                                  __bf16* __restrict__ out,
                                                  int K, int N) {
    __shared__ __bf16 tl[32][33];
    const int tid = threadIdx.x, tx = tid & 31, ty = tid >> 5;
    const int n0 = blockIdx.x * 32, k0 = blockIdx.y * 32;
#pragma unroll
    for (int j = 0; j < 4; ++j)
        tl[ty + 8 * j][tx] = (__bf16)in[(size_t)(k0 + ty + 8 * j) * N + n0 + tx];
    __syncthreads();
#pragma unroll
    for (int j = 0; j < 4; ++j)
        out[(size_t)(n0 + ty + 8 * j) * K + k0 + tx] = tl[tx][ty + 8 * j];
}

// ---------------------------------------------------------------- LayerNorm
__global__ __launch_bounds__(256) void ln_bf16(const float* __restrict__ x, int ldx,
                                               const float* __restrict__ g,
                                               const float* __restrict__ bvec,
                                               __bf16* __restrict__ out) {
    __shared__ float red[256], red2[256];
    __shared__ float mu_s, rv_s;
    const int row = blockIdx.x, tid = threadIdx.x;
    const float* xr = x + (size_t)row * ldx;
    float v[4], s = 0.f, s2 = 0.f;
#pragma unroll
    for (int j = 0; j < 4; ++j) {
        int c = tid + j * 256;
        v[j] = xr[c];
        s += v[j];
        s2 += v[j] * v[j];
    }
    red[tid] = s; red2[tid] = s2;
    __syncthreads();
    for (int off = 128; off > 0; off >>= 1) {
        if (tid < off) { red[tid] += red[tid + off]; red2[tid] += red2[tid + off]; }
        __syncthreads();
    }
    if (tid == 0) {
        float mu = red[0] * (1.f / 1024.f);
        float var = red2[0] * (1.f / 1024.f) - mu * mu;
        mu_s = mu;
        rv_s = rsqrtf(var + 1e-5f);
    }
    __syncthreads();
    const float mu = mu_s, rv = rv_s;
    __bf16* orow = out + (size_t)row * EE;
#pragma unroll
    for (int j = 0; j < 4; ++j) {
        int c = tid + j * 256;
        orow[c] = (__bf16)((v[j] - mu) * rv * g[c] + bvec[c]);
    }
}

// ---------------------------------------------------------------- WMMA GEMM
// C[M,N] = A[M,K](bf16 row-major) x Bt[N,K](bf16, pre-transposed) + bias
// MODE 0: out bf16   MODE 1: out f32 + residual f32   MODE 2: gelu -> bf16
// Block tile 128x128, K-step 64 -> 16 WMMAs per barrier pair.
template<int MODE>
__global__ __launch_bounds__(256) void gemm_bf16_k(
    const __bf16* __restrict__ A, const __bf16* __restrict__ Bt,
    const float* __restrict__ bias, float* __restrict__ outF,
    __bf16* __restrict__ outB, const float* __restrict__ resid,
    int K, int lda, int ldbt, int ldo, int ldr)
{
    __shared__ __bf16 lsA[128 * 72];   // [m][k], padded stride 72 (16B-aligned frags)
    __shared__ __bf16 lsB[128 * 72];   // [n][k], padded stride 72
    const int tid  = threadIdx.x;
    const int lane = tid & 31, wid = tid >> 5;
    const int wm = wid >> 2, wn = wid & 3;              // wave grid 2x4
    const int bm0 = blockIdx.y * 128, bn0 = blockIdx.x * 128;
    const int cl  = lane & 15;
    const int hk8 = (lane >> 4) * 8, hk16 = (lane >> 4) * 16;

    f32x8 acc[4][2];
#pragma unroll
    for (int i = 0; i < 4; ++i)
#pragma unroll
        for (int j = 0; j < 2; ++j)
#pragma unroll
            for (int r = 0; r < 8; ++r)
                acc[i][j][r] = 0.f;

    const int sr = tid >> 1, skc = (tid & 1) * 32;      // stage: 128 rows x 64 k

    for (int k0 = 0; k0 < K; k0 += 64) {
#ifndef USE_ASYNC_LDS
        bf16x8 av[4], bv[4];
#pragma unroll
        for (int j = 0; j < 4; ++j) {
            av[j] = *(const bf16x8*)&A [(size_t)(bm0 + sr) * lda  + k0 + skc + 8 * j];
            bv[j] = *(const bf16x8*)&Bt[(size_t)(bn0 + sr) * ldbt + k0 + skc + 8 * j];
        }
#endif
        if (k0 + 64 < K) {   // gfx1250 global_prefetch_b8 of next K tile
            __builtin_prefetch((const void*)&A [(size_t)(bm0 + sr) * lda  + k0 + 64 + skc], 0, 1);
            __builtin_prefetch((const void*)&Bt[(size_t)(bn0 + sr) * ldbt + k0 + 64 + skc], 0, 1);
        }
        __syncthreads();                                // prev-iter consumers done
#ifdef USE_ASYNC_LDS
#pragma unroll
        for (int j = 0; j < 4; ++j) {
            cp_async16(&lsA[sr * 72 + skc + 8 * j],
                       &A [(size_t)(bm0 + sr) * lda  + k0 + skc + 8 * j]);
            cp_async16(&lsB[sr * 72 + skc + 8 * j],
                       &Bt[(size_t)(bn0 + sr) * ldbt + k0 + skc + 8 * j]);
        }
        async_wait0();
#else
#pragma unroll
        for (int j = 0; j < 4; ++j) {
            *(bf16x8*)&lsA[sr * 72 + skc + 8 * j] = av[j];
            *(bf16x8*)&lsB[sr * 72 + skc + 8 * j] = bv[j];
        }
#endif
        __syncthreads();

#pragma unroll
        for (int s = 0; s < 2; ++s) {                   // two 32-wide K-steps
            bf16x16 af[4], bfr[2];
#pragma unroll
            for (int mt = 0; mt < 4; ++mt) {
                const __bf16* pa = &lsA[(wm * 64 + mt * 16 + cl) * 72 + s * 32 + hk8];
                af[mt] = ld16_pair(pa, pa + 16);
            }
#pragma unroll
            for (int nt = 0; nt < 2; ++nt) {
                const __bf16* pb = &lsB[(wn * 32 + nt * 16 + cl) * 72 + s * 32 + hk8];
                bfr[nt] = ld16_pair(pb, pb + 16);
            }
#pragma unroll
            for (int mt = 0; mt < 4; ++mt)
#pragma unroll
                for (int nt = 0; nt < 2; ++nt)
                    acc[mt][nt] = wmma_bf16(af[mt], bfr[nt], acc[mt][nt]);
        }
    }

    // Epilogue: lane cl = col, VGPR r = row (hk8 selects row half)
#pragma unroll
    for (int mt = 0; mt < 4; ++mt) {
        int grow0 = bm0 + wm * 64 + mt * 16 + hk8;
#pragma unroll
        for (int nt = 0; nt < 2; ++nt) {
            int gcol = bn0 + wn * 32 + nt * 16 + cl;
            float bv2 = bias[gcol];
#pragma unroll
            for (int r = 0; r < 8; ++r) {
                float v = acc[mt][nt][r] + bv2;
                int grow = grow0 + r;
                if (MODE == 0) {
                    outB[(size_t)grow * ldo + gcol] = (__bf16)v;
                } else if (MODE == 1) {
                    outF[(size_t)grow * ldo + gcol] = v + resid[(size_t)grow * ldr + gcol];
                } else {
                    float ge = 0.5f * v * (1.f + erff(v * 0.70710678118f));
                    outB[(size_t)grow * ldo + gcol] = (__bf16)ge;
                }
            }
        }
    }
}

// ---------------------------------------------------------------- Flash attention
// qkv: bf16 [8192][3072]  (q | k | v, head h at col h*64)
// One block = one (b,h) and 128 query rows; each wave owns 16 rows.
__global__ __launch_bounds__(256) void attn_k(const __bf16* __restrict__ qkv,
                                              __bf16* __restrict__ o) {
    __shared__ __bf16 lsK[32 * 64];    // [key][d]
    __shared__ __bf16 lsV[64 * 40];    // transposed [d][key], padded stride 40
    __shared__ __bf16 lsP[8 * 512];    // per-wave P tile 16x32
    const int tid = threadIdx.x, lane = tid & 31, wid = tid >> 5;
    const int qblk = blockIdx.x & 7;
    const int bh = blockIdx.x >> 3;
    const int b = bh >> 4, h = bh & 15;
    const int rowq0 = b * PP + qblk * 128 + wid * 16;
    const int cl = lane & 15, hk8 = (lane >> 4) * 8, hk16 = (lane >> 4) * 16;
    __bf16* lsPw = lsP + wid * 512;

    // Q fragments (constant across key loop)
    bf16x16 qf[2];
    const __bf16* qbase = qkv + (size_t)(rowq0 + cl) * 3072 + h * DHD;
#pragma unroll
    for (int s = 0; s < 2; ++s)
        qf[s] = ld16_pair(qbase + s * 32 + hk8, qbase + s * 32 + hk8 + 16);

    f32x8 oacc[4];
#pragma unroll
    for (int t = 0; t < 4; ++t)
#pragma unroll
        for (int r = 0; r < 8; ++r) oacc[t][r] = 0.f;
    float mrow[8], lrow[8];
#pragma unroll
    for (int r = 0; r < 8; ++r) { mrow[r] = -1e30f; lrow[r] = 0.f; }

    const int ldk = tid >> 3, ldc = (tid & 7) * 8;   // staging: key, d-offset

    for (int kc = 0; kc < PP; kc += 32) {
        const size_t krow = (size_t)(b * PP + kc + ldk) * 3072 + h * DHD;
        bf16x8 vv = *(const bf16x8*)&qkv[krow + 2 * EE + ldc];
#ifndef USE_ASYNC_LDS
        bf16x8 kv = *(const bf16x8*)&qkv[krow + EE + ldc];
#endif
        __syncthreads();
#ifdef USE_ASYNC_LDS
        cp_async16(&lsK[ldk * 64 + ldc], &qkv[krow + EE + ldc]);
#else
        *(bf16x8*)&lsK[ldk * 64 + ldc] = kv;
#endif
#pragma unroll
        for (int j = 0; j < 8; ++j) lsV[(ldc + j) * 40 + ldk] = vv[j];
#ifdef USE_ASYNC_LDS
        async_wait0();
#endif
        __syncthreads();

        // S = Q K^T
        f32x8 s0, s1;
#pragma unroll
        for (int r = 0; r < 8; ++r) { s0[r] = 0.f; s1[r] = 0.f; }
#pragma unroll
        for (int s = 0; s < 2; ++s) {
            const __bf16* p0 = &lsK[(cl) * 64 + s * 32 + hk8];
            const __bf16* p1 = &lsK[(16 + cl) * 64 + s * 32 + hk8];
            bf16x16 bk0 = ld16_pair(p0, p0 + 16);
            bf16x16 bk1 = ld16_pair(p1, p1 + 16);
            s0 = wmma_bf16(qf[s], bk0, s0);
            s1 = wmma_bf16(qf[s], bk1, s1);
        }

        // Online softmax per row (row = hk8+r, col = cl; reduce over 16 lanes)
#pragma unroll
        for (int r = 0; r < 8; ++r) {
            float a0 = s0[r] * ATTN_SCALE, a1 = s1[r] * ATTN_SCALE;
            float rmax = fmaxf(a0, a1);
            rmax = fmaxf(rmax, __shfl_xor(rmax, 1, 16));
            rmax = fmaxf(rmax, __shfl_xor(rmax, 2, 16));
            rmax = fmaxf(rmax, __shfl_xor(rmax, 4, 16));
            rmax = fmaxf(rmax, __shfl_xor(rmax, 8, 16));
            float mnew  = fmaxf(mrow[r], rmax);
            float alpha = __expf(mrow[r] - mnew);
            float p0 = __expf(a0 - mnew), p1 = __expf(a1 - mnew);
            float rs = p0 + p1;
            rs += __shfl_xor(rs, 1, 16);
            rs += __shfl_xor(rs, 2, 16);
            rs += __shfl_xor(rs, 4, 16);
            rs += __shfl_xor(rs, 8, 16);
            lrow[r] = lrow[r] * alpha + rs;
            mrow[r] = mnew;
#pragma unroll
            for (int t = 0; t < 4; ++t) oacc[t][r] *= alpha;
            int prow = hk8 + r;
            lsPw[prow * 32 + cl]      = (__bf16)p0;
            lsPw[prow * 32 + 16 + cl] = (__bf16)p1;
        }
        __syncthreads();

        // O += P V : P is exactly one 16x32 A fragment
        const __bf16* pp = &lsPw[cl * 32 + hk8];
        bf16x16 pf = ld16_pair(pp, pp + 16);
#pragma unroll
        for (int t = 0; t < 4; ++t) {
            const __bf16* pv = &lsV[(t * 16 + cl) * 40 + hk8];
            bf16x16 vf = ld16_pair(pv, pv + 16);
            oacc[t] = wmma_bf16(pf, vf, oacc[t]);
        }
    }

    // Normalize and write o (bf16, [8192][1024])
#pragma unroll
    for (int t = 0; t < 4; ++t)
#pragma unroll
        for (int r = 0; r < 8; ++r) {
            int grow = rowq0 + hk8 + r;
            o[(size_t)grow * EE + h * DHD + t * 16 + cl] = (__bf16)(oacc[t][r] / lrow[r]);
        }
}

// ---------------------------------------------------------------- pad column
__global__ void zerotail_k(float* __restrict__ out) {
    int i = blockIdx.x * 256 + threadIdx.x;      // 8192 rows exactly
    out[(size_t)i * (EE + 1) + EE] = 0.f;
}

// ================================================================ launch
extern "C" void kernel_launch(void* const* d_in, const int* in_sizes, int n_in,
                              void* d_out, int out_size, void* d_ws, size_t ws_size,
                              hipStream_t stream) {
    (void)in_sizes; (void)n_in; (void)out_size; (void)ws_size;
    const float* x    = (const float*)d_in[0];   // (8,1024,1025)
    const float* ln1g = (const float*)d_in[1];
    const float* ln1b = (const float*)d_in[2];
    const float* wqkv = (const float*)d_in[3];   // (1024,3072)
    const float* bqkv = (const float*)d_in[4];
    const float* wo   = (const float*)d_in[5];   // (1024,1024)
    const float* bo   = (const float*)d_in[6];
    const float* ln2g = (const float*)d_in[7];
    const float* ln2b = (const float*)d_in[8];
    const float* w1   = (const float*)d_in[9];   // (1024,4096)
    const float* b1   = (const float*)d_in[10];
    const float* w2   = (const float*)d_in[11];  // (4096,1024)
    const float* b2   = (const float*)d_in[12];
    float* out = (float*)d_out;

    char* ws = (char*)d_ws;
    __bf16* wqkvT = (__bf16*)(ws);                  // [3072][1024] 6,291,456 B
    __bf16* woT   = (__bf16*)(ws + 6291456);        // [1024][1024] 2,097,152 B
    __bf16* w1T   = (__bf16*)(ws + 8388608);        // [4096][1024] 8,388,608 B
    __bf16* w2T   = (__bf16*)(ws + 16777216);       // [1024][4096] 8,388,608 B
    __bf16* xn_b  = (__bf16*)(ws + 25165824);       // 16,777,216 B
    __bf16* qkv_b = (__bf16*)(ws + 41943040);       // 50,331,648 B
    __bf16* o_b   = (__bf16*)(ws + 92274688);       // 16,777,216 B
    float*  x2_f  = (float*) (ws + 109051904);      // 33,554,432 B
    __bf16* h_b   = (__bf16*)(ws + 142606336);      // 16,777,216 B
    __bf16* h1_b  = (__bf16*)(ws + 159383552);      // 67,108,864 B

    // 1) weights -> bf16, transposed to [N][K] for vectorized LDS staging
    cvt_t_bf16<<<dim3(96, 32),  256, 0, stream>>>(wqkv, wqkvT, EE,   3 * EE);
    cvt_t_bf16<<<dim3(32, 32),  256, 0, stream>>>(wo,   woT,   EE,   EE);
    cvt_t_bf16<<<dim3(128, 32), 256, 0, stream>>>(w1,   w1T,   EE,   MLPD);
    cvt_t_bf16<<<dim3(32, 128), 256, 0, stream>>>(w2,   w2T,   MLPD, EE);

    // 2) LN1 on x[..., :E] (row stride 1025)
    ln_bf16<<<ROWS, 256, 0, stream>>>(x, EE + 1, ln1g, ln1b, xn_b);

    // 3) qkv = xn @ wqkv + bqkv -> bf16 [8192][3072]
    gemm_bf16_k<0><<<dim3(24, 64), 256, 0, stream>>>(
        xn_b, wqkvT, bqkv, nullptr, qkv_b, nullptr,
        /*K=*/EE, /*lda=*/EE, /*ldbt=*/EE, /*ldo=*/3 * EE, /*ldr=*/0);

    // 4) flash attention -> o_b [8192][1024]
    attn_k<<<BB * HH * (PP / 128), 256, 0, stream>>>(qkv_b, o_b);

    // 5) x2 = o @ wo + bo + x_in -> f32 [8192][1024]
    gemm_bf16_k<1><<<dim3(8, 64), 256, 0, stream>>>(
        o_b, woT, bo, x2_f, nullptr, x,
        /*K=*/EE, /*lda=*/EE, /*ldbt=*/EE, /*ldo=*/EE, /*ldr=*/EE + 1);

    // 6) LN2
    ln_bf16<<<ROWS, 256, 0, stream>>>(x2_f, EE, ln2g, ln2b, h_b);

    // 7) h1 = gelu(h @ w1 + b1) -> bf16 [8192][4096]
    gemm_bf16_k<2><<<dim3(32, 64), 256, 0, stream>>>(
        h_b, w1T, b1, nullptr, h1_b, nullptr,
        /*K=*/EE, /*lda=*/EE, /*ldbt=*/EE, /*ldo=*/MLPD, /*ldr=*/0);

    // 8) out = h1 @ w2 + b2 + x2 -> f32 straight into d_out (stride 1025)
    gemm_bf16_k<1><<<dim3(8, 64), 256, 0, stream>>>(
        h1_b, w2T, b2, out, nullptr, x2_f,
        /*K=*/MLPD, /*lda=*/MLPD, /*ldbt=*/MLPD, /*ldo=*/EE + 1, /*ldr=*/EE);

    // 9) zero pad column 1024
    zerotail_k<<<ROWS / 256, 256, 0, stream>>>(out);
}
